// AttentiveStateMLP_64682207478119
// MI455X (gfx1250) — matrix operand order repaired and checked by reading, not compiled, not run on hardware
//
#include <hip/hip_runtime.h>
#include <hip/hip_bf16.h>
#include <math.h>

typedef __attribute__((ext_vector_type(16))) _Float16 v16h;
typedef __attribute__((ext_vector_type(8)))  _Float16 v8h;
typedef __attribute__((ext_vector_type(2)))  _Float16 v2h;
typedef __attribute__((ext_vector_type(8)))  float    v8f;
typedef __attribute__((ext_vector_type(4)))  unsigned int v4u;
typedef __attribute__((ext_vector_type(8)))  int v8i;
typedef __attribute__((ext_vector_type(4)))  int v4i;

namespace {

constexpr int kB       = 131072;
constexpr int kThreads = 256;          // 8 waves; 2 waves cooperate per 16-sample tile
constexpr int kPairs   = 4;
constexpr int kSampPerPair  = 16;
constexpr int kSampPerBlock = kPairs * kSampPerPair; // 64

// ---------------- LDS layout (bytes) ----------------
constexpr int OFF_WQKV = 0;       // f16 swizzled [K=64][N=192] KT=2  24576
constexpr int OFF_WO   = 24576;   // [64][64]  KT=2                    8192
constexpr int OFF_WP   = 32768;   // [64][128] KT=2                   16384
constexpr int OFF_W1P  = 49152;   // [32][64]                          4096
constexpr int OFF_W1O  = 53248;   // [32][32]                          2048
constexpr int OFF_W1M  = 55296;   // [32][16]                          1024
constexpr int OFF_W1G  = 56320;   // [32][16]                          1024
constexpr int OFF_W1S  = 57344;   // [32][16]                          1024
constexpr int OFF_P_P  = 58368;   // [64][64] KT=2                     8192
constexpr int OFF_P_O  = 66560;   // [32][64]                          4096
constexpr int OFF_P_M  = 70656;   // [32][64]                          4096
constexpr int OFF_P_G  = 74752;   // [32][64]                          4096
constexpr int OFF_P_S  = 78848;   // [32][64]                          4096
constexpr int OFF_BIAS = 82944;   // 976 f32                           3904
constexpr int OFF_ARENA= 87040;
constexpr int ARENA_SZ = 53248;   // per sample-pair tile
// arena (per tile); regions reused across pipeline phases
constexpr int AR_XF   = 0;        // f16 [16][160]                (phase 1)
constexpr int AR_XRAW = 5120;     // f32 [16][58] packed, 3712 B  (phase 1, TDM dest)
constexpr int AR_FT   = 10240;    // f16 [16][192]                (phase 1)
constexpr int AR_QKV  = 0;        // f16 [80][192]                (phase 2)
constexpr int AR_HH   = 0;        // f32 [80][64]                 (phase 3)
constexpr int AR_TOK  = 30720;    // f16 [80][64]   row = s*5+t
constexpr int AR_CTX  = 40960;    // f16 [80][64]
constexpr int AR_POOL = 51200;    // f16 [16][64]
constexpr int LDS_BYTES = OFF_ARENA + kPairs * ARENA_SZ; // 300032 <= 320K

// bias region float indices
constexpr int BQKV = 0;    // 192
constexpr int BO   = 192;  // 64
constexpr int BP   = 256;  // 128
constexpr int B1   = 384;  // 144
constexpr int PB   = 528;  // 320
constexpr int GM   = 848;  // 64
constexpr int BT   = 912;  // 64

__device__ __forceinline__ v8f wmma_step(v16h a, v16h b, v8f c) {
  return __builtin_amdgcn_wmma_f32_16x16x32_f16(false, a, false, b, (short)0, c,
                                                false, false);
}

__device__ __forceinline__ v8f csplat(float bv) {
  return (v8f){bv, bv, bv, bv, bv, bv, bv, bv};
}

// A-fragment (16x32 f16) per ISA layout: two contiguous 16B reads per lane.
__device__ __forceinline__ v16h load_a_frag(const _Float16* lds, int row0,
                                            int ldm, int k0, int lane) {
  int m = lane & 15, hi = lane >> 4;
  const _Float16* p = lds + (size_t)(row0 + m) * ldm + k0;
  union { v16h v; v8h h[2]; } u;
  u.h[0] = *(const v8h*)(p + hi * 8);
  u.h[1] = *(const v8h*)(p + 16 + hi * 8);
  return u.v;
}

// B-fragments pre-swizzled: one contiguous 32B read per lane.
__device__ __forceinline__ v16h load_b_frag(const _Float16* base, int kt,
                                            int nt, int KT, int lane) {
  return *(const v16h*)(base + (((size_t)(nt * KT + kt) * 32 + lane) << 4));
}

__device__ __forceinline__ void store_b_elem(_Float16* base, int k, int n,
                                             int KT, _Float16 v) {
  int kt = k >> 5, kin = k & 31, nt = n >> 4, nin = n & 15;
  int lane = (((kin >> 3) & 1) << 4) | nin;
  int e = (kin & 7) | ((kin >> 4) << 3);
  base[((((size_t)nt * KT + kt) * 32 + lane) << 4) + e] = v;
}

template <bool RELU>
__device__ __forceinline__ void store_c_f16(_Float16* dst, int rbase,
                                            int rstride, int ldn, v8f c,
                                            int lane) {
  int n = lane & 15, mb = (lane >> 4) * 8;
#pragma unroll
  for (int r = 0; r < 8; ++r) {
    float v = c[r];
    if (RELU) v = v > 0.f ? v : 0.f;
    dst[(size_t)(rbase + (mb + r) * rstride) * ldn + n] = (_Float16)v;
  }
}

// 16 consecutive f16 (one attention row) as two b128 LDS reads.
union Row16 { v8h h8[2]; v2h h2[8]; };
__device__ __forceinline__ Row16 load_row16(const _Float16* p) {
  Row16 r;
  r.h8[0] = *(const v8h*)p;
  r.h8[1] = *(const v8h*)(p + 8);
  return r;
}

struct Params {
  const float* in[29];
  float* out;
};

__global__ __launch_bounds__(kThreads, 1) void attn_state_mlp(Params P) {
  extern __shared__ char smem[];
  const int tid = threadIdx.x, lane = tid & 31, wave = tid >> 5;
  const int pairId = wave >> 1, sub = wave & 1;
  const int tid2 = sub * 32 + lane;  // 0..63 within the cooperating wave pair

  const float* x      = P.in[0];
  const float* W_phys = P.in[1];  const float* b_phys = P.in[2];
  const float* W_obj  = P.in[3];  const float* b_obj  = P.in[4];
  const float* W_mine = P.in[5];  const float* b_mine = P.in[6];
  const float* W_prog = P.in[7];  const float* b_prog = P.in[8];
  const float* W_seq  = P.in[9];  const float* b_seq  = P.in[10];
  const float* P_phys = P.in[11]; const float* pb_phys= P.in[12];
  const float* P_obj  = P.in[13]; const float* pb_obj = P.in[14];
  const float* P_mine = P.in[15]; const float* pb_mine= P.in[16];
  const float* P_prog = P.in[17]; const float* pb_prog= P.in[18];
  const float* P_seq  = P.in[19]; const float* pb_seq = P.in[20];
  const float* Wqkv   = P.in[21]; const float* bqkv   = P.in[22];
  const float* Wo     = P.in[23]; const float* bo     = P.in[24];
  const float* gamma  = P.in[25]; const float* beta   = P.in[26];
  const float* Wp     = P.in[27]; const float* bp     = P.in[28];

  _Float16* WQ  = (_Float16*)(smem + OFF_WQKV);
  _Float16* WOB = (_Float16*)(smem + OFF_WO);
  _Float16* WPB = (_Float16*)(smem + OFF_WP);
  float*    BIAS= (float*)(smem + OFF_BIAS);

  // ---------------- cooperative weight preload (f16, swizzled) --------------
  for (int i = tid; i < 64 * 192; i += kThreads) {
    int k = i & 63, n = i >> 6;
    store_b_elem(WQ, k, n, 2, (_Float16)Wqkv[n * 64 + k]);
  }
  for (int i = tid; i < 64 * 64; i += kThreads) {
    int k = i & 63, n = i >> 6;
    store_b_elem(WOB, k, n, 2, (_Float16)Wo[n * 64 + k]);
  }
  for (int i = tid; i < 64 * 128; i += kThreads) {
    int k = i & 63, n = i >> 6;
    store_b_elem(WPB, k, n, 2, (_Float16)Wp[n * 64 + k]);
  }
  for (int i = tid; i < 32 * 64; i += kThreads) {
    int k = i & 31, n = i >> 5;
    store_b_elem((_Float16*)(smem + OFF_W1P), k, n, 1,
                 (_Float16)(k < 29 ? W_phys[n * 29 + k] : 0.f));
  }
  for (int i = tid; i < 32 * 32; i += kThreads) {
    int k = i & 31, n = i >> 5;
    store_b_elem((_Float16*)(smem + OFF_W1O), k, n, 1,
                 (_Float16)(k < 15 ? W_obj[n * 15 + k] : 0.f));
  }
  for (int i = tid; i < 32 * 16; i += kThreads) {
    int k = i & 31, n = i >> 5;
    store_b_elem((_Float16*)(smem + OFF_W1M), k, n, 1,
                 (_Float16)(k < 8 ? W_mine[n * 8 + k] : 0.f));
  }
  for (int i = tid; i < 32 * 16; i += kThreads) {
    int k = i & 31, n = i >> 5;
    store_b_elem((_Float16*)(smem + OFF_W1G), k, n, 1,
                 (_Float16)(k < 3 ? W_prog[n * 3 + k] : 0.f));
  }
  for (int i = tid; i < 32 * 16; i += kThreads) {
    int k = i & 31, n = i >> 5;
    store_b_elem((_Float16*)(smem + OFF_W1S), k, n, 1,
                 (_Float16)(k < 3 ? W_seq[n * 3 + k] : 0.f));
  }
  for (int i = tid; i < 64 * 64; i += kThreads) {
    int k = i & 63, n = i >> 6;
    store_b_elem((_Float16*)(smem + OFF_P_P), k, n, 2,
                 (_Float16)P_phys[n * 64 + k]);
  }
  for (int i = tid; i < 32 * 64; i += kThreads) {
    int k = i & 31, n = i >> 5;
    store_b_elem((_Float16*)(smem + OFF_P_O), k, n, 1,
                 (_Float16)P_obj[n * 32 + k]);
  }
  for (int i = tid; i < 32 * 64; i += kThreads) {
    int k = i & 31, n = i >> 5;
    store_b_elem((_Float16*)(smem + OFF_P_M), k, n, 1,
                 (_Float16)(k < 16 ? P_mine[n * 16 + k] : 0.f));
  }
  for (int i = tid; i < 32 * 64; i += kThreads) {
    int k = i & 31, n = i >> 5;
    store_b_elem((_Float16*)(smem + OFF_P_G), k, n, 1,
                 (_Float16)(k < 16 ? P_prog[n * 16 + k] : 0.f));
  }
  for (int i = tid; i < 32 * 64; i += kThreads) {
    int k = i & 31, n = i >> 5;
    store_b_elem((_Float16*)(smem + OFF_P_S), k, n, 1,
                 (_Float16)(k < 16 ? P_seq[n * 16 + k] : 0.f));
  }
  for (int i = tid; i < 976; i += kThreads) {
    float v;
    if      (i < 192) v = bqkv[i];
    else if (i < 256) v = bo[i - 192];
    else if (i < 384) v = bp[i - 256];
    else if (i < 448) v = b_phys[i - 384];
    else if (i < 480) v = b_obj[i - 448];
    else if (i < 496) v = b_mine[i - 480];
    else if (i < 512) v = b_prog[i - 496];
    else if (i < 528) v = b_seq[i - 512];
    else if (i < 592) v = pb_phys[i - 528];
    else if (i < 656) v = pb_obj[i - 592];
    else if (i < 720) v = pb_mine[i - 656];
    else if (i < 784) v = pb_prog[i - 720];
    else if (i < 848) v = pb_seq[i - 784];
    else if (i < 912) v = gamma[i - 848];
    else              v = beta[i - 912];
    BIAS[i] = v;
  }

  char* wb = smem + OFF_ARENA + pairId * ARENA_SZ;
  _Float16* XF   = (_Float16*)(wb + AR_XF);     // [16][160]
  float*    XRAW = (float*)   (wb + AR_XRAW);   // [16][58] packed
  _Float16* FT   = (_Float16*)(wb + AR_FT);     // [16][192]
  _Float16* QKVp = (_Float16*)(wb + AR_QKV);    // [80][192]
  _Float16* TOK  = (_Float16*)(wb + AR_TOK);    // [80][64]
  _Float16* CTX  = (_Float16*)(wb + AR_CTX);    // [80][64]
  float*    HH   = (float*)   (wb + AR_HH);     // [80][64]
  _Float16* POOL = (_Float16*)(wb + AR_POOL);   // [16][64]

  const size_t sbase = (size_t)blockIdx.x * kSampPerBlock + pairId * kSampPerPair;
  const int n16 = lane & 15;

  // ---- TDM: each wave DMA-loads its 8 rows of x (58 f32) into XRAW --------
  {
    unsigned lds_addr = (unsigned)(size_t)(XRAW + (size_t)sub * 8 * 58);
    unsigned long long ga =
        (unsigned long long)(size_t)(x + (sbase + sub * 8) * 58);
    v4u g0;
    g0[0] = 1u;                       // count=1, user descriptor
    g0[1] = lds_addr;                 // lds_addr
    g0[2] = (unsigned)ga;             // global_addr[31:0]
    g0[3] = (unsigned)((ga >> 32) & 0x01FFFFFFu) | (2u << 30);  // addr hi | type=2
    v8i g1;
    g1[0] = 0x20000;                  // data_size=4B, no pad/iterate/multicast
    g1[1] = (58 << 16);               // tensor_dim0 = 58
    g1[2] = (8 << 16);                // tensor_dim1 = 8
    g1[3] = (58 << 16);               // tile_dim0 = 58
    g1[4] = 8;                        // tile_dim1 = 8, tile_dim2 = 0
    g1[5] = 58;                       // tensor_dim0_stride = 58
    g1[6] = 0;
    g1[7] = 0;
    v4i z4 = {0, 0, 0, 0};
    v8i z8 = {0, 0, 0, 0, 0, 0, 0, 0};
    __builtin_amdgcn_tensor_load_to_lds(g0, g1, z4, z4, z8, 0);
    __builtin_amdgcn_s_wait_tensorcnt(0);
  }
  __syncthreads();

  // stage x -> K-padded f16 layout (zero pads), zero FT
  {
    const int xw[5] = {29, 15, 8, 3, 3};
    const int xb[5] = {0, 29, 44, 52, 55};
    for (int i = tid2; i < 16 * 160; i += 64) {
      int r = i / 160, c = i % 160, g = c >> 5, ci = c & 31;
      float v = (ci < xw[g]) ? XRAW[r * 58 + xb[g] + ci] : 0.f;
      XF[r * 160 + c] = (_Float16)v;
    }
    for (int i = tid2; i < 16 * 192; i += 64) FT[i] = (_Float16)0.f;
  }
  __syncthreads();

  // GEMM1: feature MLPs + relu (bias folded into C init)
  if (sub == 0) {
    v16h a = load_a_frag(XF, 0, 160, 0, lane);
    for (int nt = 0; nt < 4; ++nt) {
      v8f c = csplat(BIAS[B1 + nt * 16 + n16]);
      c = wmma_step(a, load_b_frag((const _Float16*)(smem + OFF_W1P), 0, nt, 1, lane), c);
      store_c_f16<true>(FT + nt * 16, 0, 1, 192, c, lane);
    }
  } else {
    struct G { int k0, woff, nt, c0, b0; };
    const G gs[4] = {{32,  OFF_W1O, 2, 64,  B1 + 64},
                     {64,  OFF_W1M, 1, 96,  B1 + 96},
                     {96,  OFF_W1G, 1, 128, B1 + 112},
                     {128, OFF_W1S, 1, 160, B1 + 128}};
#pragma unroll
    for (int g = 0; g < 4; ++g) {
      v16h a = load_a_frag(XF, 0, 160, gs[g].k0, lane);
      const _Float16* w = (const _Float16*)(smem + gs[g].woff);
      for (int nt = 0; nt < gs[g].nt; ++nt) {
        v8f c = csplat(BIAS[gs[g].b0 + nt * 16 + n16]);
        c = wmma_step(a, load_b_frag(w, 0, nt, 1, lane), c);
        store_c_f16<true>(FT + gs[g].c0 + nt * 16, 0, 1, 192, c, lane);
      }
    }
  }
  __syncthreads();

  // GEMM2: token projections -> TOK rows s*5+t (each wave does 2 of 4 n-tiles)
  {
    struct G { int k0, woff, kt, t, pb; };
    const G gs[5] = {{0,   OFF_P_P, 2, 0, PB + 0},
                     {64,  OFF_P_O, 1, 1, PB + 64},
                     {96,  OFF_P_M, 1, 2, PB + 128},
                     {128, OFF_P_G, 1, 3, PB + 192},
                     {160, OFF_P_S, 1, 4, PB + 256}};
#pragma unroll
    for (int g = 0; g < 5; ++g) {
      const _Float16* w = (const _Float16*)(smem + gs[g].woff);
      v16h a0 = load_a_frag(FT, 0, 192, gs[g].k0, lane);
      v16h a1 = (gs[g].kt == 2) ? load_a_frag(FT, 0, 192, gs[g].k0 + 32, lane)
                                : a0;
#pragma unroll
      for (int j = 0; j < 2; ++j) {
        int nt = sub * 2 + j;
        v8f c = csplat(BIAS[gs[g].pb + nt * 16 + n16]);
        c = wmma_step(a0, load_b_frag(w, 0, nt, gs[g].kt, lane), c);
        if (gs[g].kt == 2)
          c = wmma_step(a1, load_b_frag(w, 1, nt, 2, lane), c);
        store_c_f16<false>(TOK + nt * 16, gs[g].t, 5, 64, c, lane);
      }
    }
  }
  __syncthreads();

  // GEMM3: QKV [80x64]@[64x192] (each wave does 6 of 12 n-tiles per m-tile)
  for (int mt = 0; mt < 5; ++mt) {
    v16h a0 = load_a_frag(TOK, mt * 16, 64, 0, lane);
    v16h a1 = load_a_frag(TOK, mt * 16, 64, 32, lane);
    for (int j = 0; j < 6; ++j) {
      int nt = sub * 6 + j;
      v8f c = csplat(BIAS[BQKV + nt * 16 + n16]);
      c = wmma_step(a0, load_b_frag(WQ, 0, nt, 2, lane), c);
      c = wmma_step(a1, load_b_frag(WQ, 1, nt, 2, lane), c);
      store_c_f16<false>(QKVp + nt * 16, mt * 16, 1, 192, c, lane);
    }
  }
  __syncthreads();

  // attention: one (sample,head) pair per lane across the wave pair; packed f16
  {
    int s = tid2 >> 2, h = tid2 & 3;
    const _Float16* qb = QKVp + (size_t)s * 5 * 192 + h * 16;
    for (int qt = 0; qt < 5; ++qt) {
      Row16 qr = load_row16(qb + qt * 192);
      float sc[5], mx = -3e38f;
      for (int kt = 0; kt < 5; ++kt) {
        Row16 kr = load_row16(qb + 64 + kt * 192);
        v2h acc = {};
#pragma unroll
        for (int j = 0; j < 8; ++j) acc = qr.h2[j] * kr.h2[j] + acc;
        float a = ((float)acc[0] + (float)acc[1]) * 0.25f;  // 1/sqrt(16)
        sc[kt] = a;
        mx = a > mx ? a : mx;
      }
      float den = 0.f;
      for (int kt = 0; kt < 5; ++kt) { sc[kt] = __expf(sc[kt] - mx); den += sc[kt]; }
      float inv = 1.f / den;
      v2h cacc[8];
#pragma unroll
      for (int j = 0; j < 8; ++j) cacc[j] = (v2h){};
      for (int kt = 0; kt < 5; ++kt) {
        Row16 vr = load_row16(qb + 128 + kt * 192);
        _Float16 ph = (_Float16)(sc[kt] * inv);
        v2h pv = {ph, ph};
#pragma unroll
        for (int j = 0; j < 8; ++j) cacc[j] = pv * vr.h2[j] + cacc[j];
      }
      _Float16* cp = CTX + (size_t)(s * 5 + qt) * 64 + h * 16;
#pragma unroll
      for (int j = 0; j < 8; ++j) *(v2h*)(cp + 2 * j) = cacc[j];
    }
  }
  __syncthreads();

  // GEMM4: Wo + bias + residual -> HH f32 (each wave: 2 of 4 n-tiles)
  for (int mt = 0; mt < 5; ++mt) {
    v16h a0 = load_a_frag(CTX, mt * 16, 64, 0, lane);
    v16h a1 = load_a_frag(CTX, mt * 16, 64, 32, lane);
#pragma unroll
    for (int j = 0; j < 2; ++j) {
      int nt = sub * 2 + j;
      v8f c = csplat(BIAS[BO + nt * 16 + n16]);
      c = wmma_step(a0, load_b_frag(WOB, 0, nt, 2, lane), c);
      c = wmma_step(a1, load_b_frag(WOB, 1, nt, 2, lane), c);
      int mb = (lane >> 4) * 8;
#pragma unroll
      for (int r = 0; r < 8; ++r) {
        int row = mt * 16 + mb + r, col = nt * 16 + n16;
        HH[(size_t)row * 64 + col] = c[r] + (float)TOK[(size_t)row * 64 + col];
      }
    }
  }
  __syncthreads();

  // layernorm per token-row, then mean-pool over 5 tokens
  for (int r = tid2; r < 80; r += 64) {
    float* hr = HH + (size_t)r * 64;
    float mu = 0.f;
    for (int c = 0; c < 64; ++c) mu += hr[c];
    mu *= 0.015625f;
    float var = 0.f;
    for (int c = 0; c < 64; ++c) { float d = hr[c] - mu; var += d * d; }
    var *= 0.015625f;
    float is = rsqrtf(var + 1e-5f);
    for (int c = 0; c < 64; ++c)
      hr[c] = (hr[c] - mu) * is * BIAS[GM + c] + BIAS[BT + c];
  }
  __syncthreads();
  for (int i = tid2; i < 16 * 64; i += 64) {
    int s = i >> 6, c = i & 63;
    float a = 0.f;
    for (int t = 0; t < 5; ++t) a += HH[(size_t)(s * 5 + t) * 64 + c];
    POOL[(size_t)s * 64 + c] = (_Float16)(a * 0.2f);
  }
  __syncthreads();

  // GEMM5: Wp + relu -> global out [B][128] (each wave: 4 of 8 n-tiles)
  {
    v16h a0 = load_a_frag(POOL, 0, 64, 0, lane);
    v16h a1 = load_a_frag(POOL, 0, 64, 32, lane);
    for (int j = 0; j < 4; ++j) {
      int nt = sub * 4 + j;
      v8f c = csplat(BIAS[BP + nt * 16 + n16]);
      c = wmma_step(a0, load_b_frag(WPB, 0, nt, 2, lane), c);
      c = wmma_step(a1, load_b_frag(WPB, 1, nt, 2, lane), c);
      int mb = (lane >> 4) * 8;
#pragma unroll
      for (int r = 0; r < 8; ++r) {
        float v = c[r];
        v = v > 0.f ? v : 0.f;
        P.out[(sbase + mb + r) * 128 + nt * 16 + n16] = v;
      }
    }
  }
}

}  // namespace

extern "C" void kernel_launch(void* const* d_in, const int* in_sizes, int n_in,
                              void* d_out, int out_size, void* d_ws,
                              size_t ws_size, hipStream_t stream) {
  (void)in_sizes; (void)n_in; (void)out_size; (void)d_ws; (void)ws_size;
  Params P;
  for (int i = 0; i < 29; ++i) P.in[i] = (const float*)d_in[i];
  P.out = (float*)d_out;
  dim3 grid(kB / kSampPerBlock), block(kThreads);
  hipLaunchKernelGGL(attn_state_mlp, grid, block, LDS_BYTES, stream, P);
}
